// TrajGRUCell_49555332661459
// MI455X (gfx1250) — matrix operand order repaired and checked by reading, not compiled, not run on hardware
//
#include <hip/hip_runtime.h>

// ---------------------------------------------------------------------------
// TrajGRU cell forward for MI455X (gfx1250, wave32, WMMA).
//
// Exact-math simplifications vs reference:
//  * newhid = zt*h~ + (1-zt)*h~ == h~  -> Wxz conv and temp_zt einsum dead.
//  * conv(x,gx)+conv(h,gh) == conv(concat(x,h), concat(gx,gh) over Cin).
//  * Wxr/Wxh convs share im2col -> one GEMM N=128; Whr/Whh einsums share
//    warped A -> one GEMM K=320,N=128.
// All matmul work uses v_wmma_f32_16x16x32_f16 (f16 data, f32 accum).
// All bulk stores are LDS-staged so global writes are contiguous 16B chunks.
// ---------------------------------------------------------------------------

#define Bn 8
#define Cn 64
#define Hn 128
#define Wn 128
#define HWn (Hn * Wn)
#define Ln 5

typedef __attribute__((ext_vector_type(16))) _Float16 v16h;
typedef __attribute__((ext_vector_type(8)))  _Float16 v8h;
typedef __attribute__((ext_vector_type(8)))  float    v8f;

static __device__ __forceinline__ v8f vzero8() {
  v8f z;
#pragma unroll
  for (int i = 0; i < 8; ++i) z[i] = 0.0f;
  return z;
}

static __device__ __forceinline__ v8f wmma_f16f32(v16h a, v16h b, v8f c) {
  return __builtin_amdgcn_wmma_f32_16x16x32_f16(
      /*neg_a=*/false, a, /*neg_b=*/false, b,
      /*c_mod=*/(short)0, c, /*reuse_a=*/false, /*reuse_b=*/false);
}

// A fragment (16x32 f16). Lane l<16 holds row M=l, K=0..7 & 16..23; lane>=16
// holds M=l-16, K=8..15 & 24..31. Caller passes pointer to this lane's first
// channel chunk (channels kbase + khalf*8); second chunk is +16 channels.
static __device__ __forceinline__ v16h load_a_cached(const _Float16* p, bool valid) {
  union { v16h v; v8h h[2]; } A;
  if (valid) {
    A.h[0] = *(const v8h*)p;
    A.h[1] = *(const v8h*)(p + 16);
  } else {
#pragma unroll
    for (int i = 0; i < 16; ++i) A.v[i] = (_Float16)0.0f;
  }
  return A.v;
}

// B fragment (32x16 f16), pre-swizzled: lane l holds column n=l&15,
// K = (l>>4)*8+0..7 & 16+(l>>4)*8+0..7 as 16 contiguous f16 (32B) at tile + l*16.
static __device__ __forceinline__ v16h load_b_packed(const _Float16* tile, int lane) {
  union { v16h v; v8h h[2]; } B;
  const v8h* p = (const v8h*)(tile + lane * 16);
  B.h[0] = p[0];
  B.h[1] = p[1];
  return B.v;
}

// ---------------------------------------------------------------------------
// Pack inputx & hid_state (f32 NCHW) -> concatenated NHWC f16 [B][H][W][128]
// (ch 0..63 = inputx, 64..127 = hid_state). LDS transpose for coalescing.
// ---------------------------------------------------------------------------
__global__ void __launch_bounds__(256) pack_nhwc_kernel(
    const float* __restrict__ x, const float* __restrict__ h,
    _Float16* __restrict__ out) {
  __shared__ _Float16 tile[64 * 136];  // 64 pixels x 128ch (+8 pad)
  int tileId = blockIdx.x;
  int b = tileId / (HWn / 64);
  int pix0 = (tileId % (HWn / 64)) * 64;
  int t = threadIdx.x;
  int sub = t >> 6;   // 0..3
  int p = t & 63;     // pixel within tile
#pragma unroll 4
  for (int c = sub; c < 64; c += 4) {
    tile[p * 136 + c]      = (_Float16)x[((size_t)(b * 64 + c)) * HWn + pix0 + p];
    tile[p * 136 + 64 + c] = (_Float16)h[((size_t)(b * 64 + c)) * HWn + pix0 + p];
  }
  __syncthreads();
  // 64 pixels * 256B each = 512 int4 stores; 2 per thread, contiguous per pixel
  int4* dst = (int4*)(out + ((size_t)b * HWn + pix0) * 128);
#pragma unroll
  for (int k = 0; k < 2; ++k) {
    int i = t + k * 256;
    int pp = i >> 3, q = i & 7;
    const int4* src = (const int4*)(tile + pp * 136);
    dst[pp * 16 + q] = src[q];
  }
}

// ---------------------------------------------------------------------------
// Pack all weights into WMMA-B fragment order (f16), and bias sums.
// Segments: WCAT1 [25][4][2] tiles, WFLOW [25], WRH [25][2][8], WHRH [10][8],
// then 128 f32 biasRT. Each tile = 512 f16.
// ---------------------------------------------------------------------------
__global__ void __launch_bounds__(256) pack_weights_kernel(
    const float* __restrict__ gxw, const float* __restrict__ ghw,
    const float* __restrict__ flw,
    const float* __restrict__ wxr, const float* __restrict__ wxh,
    const float* __restrict__ whr, const float* __restrict__ whh,
    const float* __restrict__ whr_b, const float* __restrict__ whh_b,
    _Float16* __restrict__ wcat1, _Float16* __restrict__ wflow,
    _Float16* __restrict__ wrh, _Float16* __restrict__ whrh,
    float* __restrict__ biasRT) {
  const int N1 = 25 * 4 * 2 * 512;  // 102400
  const int N2 = 25 * 512;          // 12800
  const int N3 = 25 * 2 * 8 * 512;  // 204800
  const int N4 = 10 * 8 * 512;      // 40960
  int idx = blockIdx.x * 256 + threadIdx.x;
  if (idx < N1) {
    int e = idx & 511, tile = idx >> 9;
    int nb = tile & 1, kb = (tile >> 1) & 3, tap = tile >> 3;
    int l = e >> 4, j = e & 15;
    int k = kb * 32 + ((l >> 4) * 8 + (j & 7) + (j >> 3) * 16);  // ci 0..127
    int n = nb * 16 + (l & 15);             // cout 0..31
    float v = (k < 64) ? gxw[((size_t)(n * 64 + k)) * 25 + tap]
                       : ghw[((size_t)(n * 64 + (k - 64))) * 25 + tap];
    wcat1[idx] = (_Float16)v;
    return;
  }
  idx -= N1;
  if (idx < N2) {
    int e = idx & 511, tap = idx >> 9;
    int l = e >> 4, j = e & 15;
    int k = (l >> 4) * 8 + (j & 7) + (j >> 3) * 16;   // ci 0..31
    int n = l & 15;              // cout (pad 10->16 with zeros)
    float v = (n < 10) ? flw[((size_t)(n * 32 + k)) * 25 + tap] : 0.0f;
    wflow[idx] = (_Float16)v;
    return;
  }
  idx -= N2;
  if (idx < N3) {
    int e = idx & 511, tile = idx >> 9;
    int nb = tile & 7, kb = (tile >> 3) & 1, tap = tile >> 4;
    int l = e >> 4, j = e & 15;
    int k = kb * 32 + ((l >> 4) * 8 + (j & 7) + (j >> 3) * 16);  // ci 0..63
    int n = nb * 16 + (l & 15);             // 0..63 -> Wxr, 64..127 -> Wxh
    float v = (n < 64) ? wxr[((size_t)(n * 64 + k)) * 25 + tap]
                       : wxh[((size_t)((n - 64) * 64 + k)) * 25 + tap];
    wrh[idx] = (_Float16)v;
    return;
  }
  idx -= N3;
  if (idx < N4) {
    int e = idx & 511, tile = idx >> 9;
    int nb = tile & 7, kb = tile >> 3;      // kb 0..9
    int l = e >> 4, j = e & 15;
    int kk = kb * 32 + ((l >> 4) * 8 + (j & 7) + (j >> 3) * 16); // K = l5*64 + i, 0..319
    int l5 = kk >> 6, i = kk & 63;
    int n = nb * 16 + (l & 15);
    float v = (n < 64) ? whr[(size_t)l5 * 4096 + n * 64 + i]
                       : whh[(size_t)l5 * 4096 + (n - 64) * 64 + i];
    whrh[idx] = (_Float16)v;
    return;
  }
  idx -= N4;
  if (idx < 128) {
    float s = 0.0f;
    if (idx < 64) {
#pragma unroll
      for (int l = 0; l < Ln; ++l) s += whr_b[l * 64 + idx];
    } else {
#pragma unroll
      for (int l = 0; l < Ln; ++l) s += whh_b[l * 64 + (idx - 64)];
    }
    biasRT[idx] = s;
  }
}

// ---------------------------------------------------------------------------
// conv-f: 5x5, Cin=128 (x|h concat NHWC f16), Cout=32.
// f = leaky_relu(conv + gxb + ghb, 0.1) -> NHWC f16 [B][H][W][32].
// One block per (b,row); 8 waves = 4 m-groups (32 px) x 2 n-groups (16 co).
// Epilogue LDS-staged -> contiguous 16B global stores.
// ---------------------------------------------------------------------------
__global__ void __launch_bounds__(256) conv_f_kernel(
    const _Float16* __restrict__ xcat, const _Float16* __restrict__ wpack,
    const float* __restrict__ gxb, const float* __restrict__ ghb,
    _Float16* __restrict__ fout) {
  __shared__ _Float16 ftile[128 * 40];  // [pixel][32ch] stride 40 (80B rows)
  int b = blockIdx.x >> 7, y = blockIdx.x & 127;
  int wave = threadIdx.x >> 5, lane = threadIdx.x & 31;
  int mg = wave >> 1, ng = wave & 1;
  int am = lane & 15, ak = (lane >> 4) << 3;
  v8f acc0 = vzero8(), acc1 = vzero8();
  int ktap = 0;
  for (int kh = 0; kh < 5; ++kh) {
    int py = y + kh - 2;
    bool rowok = (py >= 0) & (py < Hn);
    const _Float16* rowbase =
        xcat + ((size_t)(b * Hn + (rowok ? py : 0)) * Wn) * 128;
    __builtin_prefetch(rowbase + (size_t)Wn * 128, 0, 3);
    for (int kw = 0; kw < 5; ++kw, ++ktap) {
      int px0 = mg * 32 + am + kw - 2;
      int px1 = px0 + 16;
      bool v0 = rowok & (px0 >= 0) & (px0 < Wn);
      bool v1 = rowok & (px1 >= 0) & (px1 < Wn);
      const _Float16* p0 = rowbase + (size_t)(v0 ? px0 : 0) * 128 + ak;
      const _Float16* p1 = rowbase + (size_t)(v1 ? px1 : 0) * 128 + ak;
#pragma unroll
      for (int kb = 0; kb < 4; ++kb) {
        v16h a0 = load_a_cached(p0 + kb * 32, v0);
        v16h a1 = load_a_cached(p1 + kb * 32, v1);
        v16h bf =
            load_b_packed(wpack + ((size_t)((ktap * 4 + kb) * 2 + ng)) * 512, lane);
        acc0 = wmma_f16f32(a0, bf, acc0);
        acc1 = wmma_f16f32(a1, bf, acc1);
      }
    }
  }
  int n = ng * 16 + (lane & 15);
  float bias = gxb[n] + ghb[n];
  int mb = (lane >> 4) * 8;
#pragma unroll
  for (int r = 0; r < 8; ++r) {
    int m0 = mg * 32 + mb + r;
    float f0 = acc0[r] + bias; f0 = f0 > 0.0f ? f0 : 0.1f * f0;
    ftile[m0 * 40 + n] = (_Float16)f0;
    float f1 = acc1[r] + bias; f1 = f1 > 0.0f ? f1 : 0.1f * f1;
    ftile[(m0 + 16) * 40 + n] = (_Float16)f1;
  }
  __syncthreads();
  // 128 px * 64B -> 256 threads store 32B contiguous each
  {
    int p = threadIdx.x >> 1, half = threadIdx.x & 1;
    const int4* src = (const int4*)(ftile + p * 40 + half * 16);
    int4* dst = (int4*)(fout + ((size_t)(b * Hn + y) * Wn + p) * 32 + half * 16);
    dst[0] = src[0];
    dst[1] = src[1];
  }
}

// ---------------------------------------------------------------------------
// conv-flow: 5x5, Cin=32 (f NHWC f16), Cout=10 (padded to N=16).
// flows f32 NCHW [B][10][HW]; per-lane stores are 32B contiguous already.
// ---------------------------------------------------------------------------
__global__ void __launch_bounds__(256) conv_flow_kernel(
    const _Float16* __restrict__ fin, const _Float16* __restrict__ wpack,
    const float* __restrict__ flow_b, float* __restrict__ flows) {
  int b = blockIdx.x >> 7, y = blockIdx.x & 127;
  int wave = threadIdx.x >> 5, lane = threadIdx.x & 31;
  int am = lane & 15, ak = (lane >> 4) << 3;
  v8f acc = vzero8();
  int ktap = 0;
  for (int kh = 0; kh < 5; ++kh) {
    int py = y + kh - 2;
    bool rowok = (py >= 0) & (py < Hn);
    const _Float16* rowbase = fin + ((size_t)(b * Hn + (rowok ? py : 0)) * Wn) * 32;
    for (int kw = 0; kw < 5; ++kw, ++ktap) {
      int px = wave * 16 + am + kw - 2;
      bool v = rowok & (px >= 0) & (px < Wn);
      v16h a = load_a_cached(rowbase + (size_t)(v ? px : 0) * 32 + ak, v);
      v16h bf = load_b_packed(wpack + (size_t)ktap * 512, lane);
      acc = wmma_f16f32(a, bf, acc);
    }
  }
  int n = lane & 15;
  int mb = (lane >> 4) * 8;
  if (n < 10) {
    float bias = flow_b[n];
    float4 lo, hi;
    lo.x = acc[0] + bias; lo.y = acc[1] + bias; lo.z = acc[2] + bias; lo.w = acc[3] + bias;
    hi.x = acc[4] + bias; hi.y = acc[5] + bias; hi.z = acc[6] + bias; hi.w = acc[7] + bias;
    float4* dst = (float4*)(flows + ((size_t)(b * 10 + n)) * HWn +
                            (size_t)y * Wn + wave * 16 + mb);
    dst[0] = lo;
    dst[1] = hi;
  }
}

// ---------------------------------------------------------------------------
// warp + einsum: bilinear-sample hid (f16 NHWC, channels 64..127 of xcat) at
// (grid+flow), fused directly into WMMA A fragments; GEMM K=320 (l,i), N=128
// (rt|ht). B tiles staged in LDS in two 40KB phases (ds_load_b128 fragments);
// LDS reused as epilogue staging so temps stores are contiguous 16B chunks.
// ---------------------------------------------------------------------------
static __device__ __forceinline__ void gather_corner(
    const _Float16* __restrict__ hidbase, int xi, int yi, float wgt, float* a) {
  if (wgt == 0.0f) return;
  if (xi < 0 || xi >= Wn || yi < 0 || yi >= Hn) return;  // padding_mode='zeros'
  const _Float16* p = hidbase + ((size_t)(yi * Wn + xi)) * 128;
  union { v8h h; _Float16 e[8]; } c0, c1;
  c0.h = *(const v8h*)p;
  c1.h = *(const v8h*)(p + 16);
#pragma unroll
  for (int t = 0; t < 8; ++t) {
    a[t]     += wgt * (float)c0.e[t];
    a[8 + t] += wgt * (float)c1.e[t];
  }
}

__global__ void __launch_bounds__(256) warp_gemm_kernel(
    const _Float16* __restrict__ xcat, const float* __restrict__ flows,
    const _Float16* __restrict__ wpack, const float* __restrict__ biasRT,
    _Float16* __restrict__ temps) {
  __shared__ char smem[40960];  // phase B-stage (40KB) / out-stage (34.8KB)
  int b = blockIdx.x >> 7, y = blockIdx.x & 127;
  int wave = threadIdx.x >> 5, lane = threadIdx.x & 31;
  int am = lane & 15, ak = (lane >> 4) << 3;
  int x = wave * 16 + am;
  int hw = y * Wn + x;
  v8f acc[8];
#pragma unroll
  for (int nb = 0; nb < 8; ++nb) acc[nb] = vzero8();

  const _Float16* bstage = (const _Float16*)smem;
  for (int phase = 0; phase < 2; ++phase) {
    // stage 40 tiles (kb = phase*5 .. +4) = 40KB = 2560 int4
    {
      const int4* src = (const int4*)(wpack + (size_t)phase * 5 * 8 * 512);
      int4* dst = (int4*)smem;
#pragma unroll
      for (int k = 0; k < 10; ++k) dst[threadIdx.x + k * 256] = src[threadIdx.x + k * 256];
    }
    __syncthreads();
    for (int kk = 0; kk < 5; ++kk) {
      int kb = phase * 5 + kk;
      int l5 = kb >> 1;
      int ci0 = (kb & 1) * 32 + ak;
      float sx = flows[((size_t)(b * 10 + l5 * 2)) * HWn + hw] + (float)x;
      float sy = flows[((size_t)(b * 10 + l5 * 2 + 1)) * HWn + hw] + (float)y;
      float x0f = floorf(sx), y0f = floorf(sy);
      float wx1 = sx - x0f, wy1 = sy - y0f;
      float wx0 = 1.0f - wx1, wy0 = 1.0f - wy1;
      int xi = (int)x0f, yi = (int)y0f;
      float a[16];
#pragma unroll
      for (int t = 0; t < 16; ++t) a[t] = 0.0f;
      const _Float16* hidbase = xcat + (size_t)b * HWn * 128 + 64 + ci0;
      gather_corner(hidbase, xi,     yi,     wx0 * wy0, a);
      gather_corner(hidbase, xi + 1, yi,     wx1 * wy0, a);
      gather_corner(hidbase, xi,     yi + 1, wx0 * wy1, a);
      gather_corner(hidbase, xi + 1, yi + 1, wx1 * wy1, a);
      v16h af;
#pragma unroll
      for (int t = 0; t < 16; ++t) af[t] = (_Float16)a[t];
#pragma unroll
      for (int nb = 0; nb < 8; ++nb) {
        v16h bf = load_b_packed(bstage + ((size_t)(kk * 8 + nb)) * 512, lane);
        acc[nb] = wmma_f16f32(af, bf, acc[nb]);
      }
    }
    __syncthreads();
  }
  // epilogue: stage f16 results in LDS [128 px][136], then coalesced stores
  _Float16* ostage = (_Float16*)smem;
  int mb = (lane >> 4) * 8, nl = lane & 15;
#pragma unroll
  for (int nb = 0; nb < 8; ++nb) {
    int n = nb * 16 + nl;
    float bias = biasRT[n];
#pragma unroll
    for (int r = 0; r < 8; ++r)
      ostage[(wave * 16 + mb + r) * 136 + n] = (_Float16)(acc[nb][r] + bias);
  }
  __syncthreads();
  {
    int p = threadIdx.x >> 1, half = threadIdx.x & 1;
    const int4* src = (const int4*)(ostage + p * 136 + half * 64);
    int4* dst = (int4*)(temps + ((size_t)b * HWn + (size_t)y * Wn + p) * 128 + half * 64);
#pragma unroll
    for (int q = 0; q < 4; ++q) dst[q] = src[q];
  }
}

// ---------------------------------------------------------------------------
// conv-final: 5x5, Cin=64 (inputx = channels 0..63 of xcat), Cout=128
// (0..63 convR=Wxr, 64..127 convH=Wxh). Fused epilogue via LDS f16 tile:
//   rt  = sigmoid(convR + Wxr_b + temp_rt)
//   hid = leaky_relu(convH + Wxh_b + rt*temp_ht, 0.1)
// Epilogue iterates pixel-fast so both duplicated outputs are written as
// contiguous float4 pairs (fully coalesced NCHW stores).
// ---------------------------------------------------------------------------
__global__ void __launch_bounds__(256) conv_final_kernel(
    const _Float16* __restrict__ xcat, const _Float16* __restrict__ wpack,
    const float* __restrict__ wxr_b, const float* __restrict__ wxh_b,
    const _Float16* __restrict__ temps, float* __restrict__ out) {
  __shared__ _Float16 ctile[128 * 130];  // [pixel][cout 0..127] stride 130
  int b = blockIdx.x >> 7, y = blockIdx.x & 127;
  int wave = threadIdx.x >> 5, lane = threadIdx.x & 31;
  int am = lane & 15, ak = (lane >> 4) << 3;
  v8f acc[8];
#pragma unroll
  for (int nb = 0; nb < 8; ++nb) acc[nb] = vzero8();

  int ktap = 0;
  for (int kh = 0; kh < 5; ++kh) {
    int py = y + kh - 2;
    bool rowok = (py >= 0) & (py < Hn);
    const _Float16* rowbase =
        xcat + ((size_t)(b * Hn + (rowok ? py : 0)) * Wn) * 128;
    __builtin_prefetch(rowbase + (size_t)Wn * 128, 0, 3);
    for (int kw = 0; kw < 5; ++kw, ++ktap) {
      int px = wave * 16 + am + kw - 2;
      bool v = rowok & (px >= 0) & (px < Wn);
      const _Float16* p = rowbase + (size_t)(v ? px : 0) * 128 + ak;
#pragma unroll
      for (int kb = 0; kb < 2; ++kb) {
        v16h a = load_a_cached(p + kb * 32, v);
#pragma unroll
        for (int nb = 0; nb < 8; ++nb) {
          v16h bf = load_b_packed(
              wpack + ((size_t)((ktap * 2 + kb) * 8 + nb)) * 512, lane);
          acc[nb] = wmma_f16f32(a, bf, acc[nb]);
        }
      }
    }
  }
  int mb = (lane >> 4) * 8, nl = lane & 15;
#pragma unroll
  for (int nb = 0; nb < 8; ++nb)
#pragma unroll
    for (int r = 0; r < 8; ++r)
      ctile[(wave * 16 + mb + r) * 130 + nb * 16 + nl] = (_Float16)acc[nb][r];
  __syncthreads();

  const size_t OUT2 = (size_t)Bn * Cn * HWn;  // second tuple element
  const _Float16* tpr = temps + ((size_t)b * HWn + (size_t)y * Wn) * 128;
#pragma unroll
  for (int k = 0; k < 4; ++k) {
    int u = threadIdx.x + k * 256;  // 64 c-groups x 16 pixel-groups
    int c = u >> 4, pg = u & 15;
    float br = wxr_b[c], bh = wxh_b[c];
    float4 lo, hi;
#pragma unroll
    for (int r = 0; r < 8; ++r) {
      int p = pg * 8 + r;
      float convR = (float)ctile[p * 130 + c] + br;
      float convH = (float)ctile[p * 130 + 64 + c] + bh;
      float trt = (float)tpr[p * 128 + c];
      float tht = (float)tpr[p * 128 + 64 + c];
      float rt = 1.0f / (1.0f + __expf(-(convR + trt)));
      float hv = convH + rt * tht;
      float hid = hv > 0.0f ? hv : 0.1f * hv;
      if (r < 4) ((float*)&lo)[r] = hid;
      else       ((float*)&hi)[r - 4] = hid;
    }
    size_t o = ((size_t)(b * Cn + c)) * HWn + (size_t)y * Wn + pg * 8;
    float4* d0 = (float4*)(out + o);
    d0[0] = lo; d0[1] = hi;
    float4* d1 = (float4*)(out + o + OUT2);
    d1[0] = lo; d1[1] = hi;
  }
}

// ---------------------------------------------------------------------------
extern "C" void kernel_launch(void* const* d_in, const int* in_sizes, int n_in,
                              void* d_out, int out_size, void* d_ws, size_t ws_size,
                              hipStream_t stream) {
  (void)in_sizes; (void)n_in; (void)out_size; (void)ws_size;
  const float* inputx = (const float*)d_in[0];
  const float* hid    = (const float*)d_in[1];
  const float* gxw    = (const float*)d_in[2];
  const float* gxb    = (const float*)d_in[3];
  const float* ghw    = (const float*)d_in[4];
  const float* ghb    = (const float*)d_in[5];
  const float* flw    = (const float*)d_in[6];
  const float* flb    = (const float*)d_in[7];
  // d_in[8..9] = Wxz_w / Wxz_b: provably dead (zt cancels in the combine).
  const float* wxr    = (const float*)d_in[10];
  const float* wxrb   = (const float*)d_in[11];
  const float* wxh    = (const float*)d_in[12];
  const float* wxhb   = (const float*)d_in[13];
  // d_in[14..15] = Whz_w / Whz_b: dead for the same reason.
  const float* whr    = (const float*)d_in[16];
  const float* whrb   = (const float*)d_in[17];
  const float* whh    = (const float*)d_in[18];
  const float* whhb   = (const float*)d_in[19];
  float* out = (float*)d_out;

  // Workspace layout (256B aligned)
  char* ws = (char*)d_ws;
  size_t off = 0;
  auto take = [&](size_t bytes) {
    off = (off + 255) & ~(size_t)255;
    size_t o = off;
    off += bytes;
    return o;
  };
  _Float16* XCAT  = (_Float16*)(ws + take((size_t)Bn * HWn * 128 * 2)); // 32MB
  _Float16* F16   = (_Float16*)(ws + take((size_t)Bn * HWn * 32 * 2));  //  8MB
  float*    FLOWS = (float*)   (ws + take((size_t)Bn * 10 * HWn * 4));  //  5MB
  _Float16* TEMPS = (_Float16*)(ws + take((size_t)Bn * HWn * 128 * 2)); // 32MB
  _Float16* WCAT1 = (_Float16*)(ws + take((size_t)25 * 4 * 2 * 512 * 2));
  _Float16* WFLOW = (_Float16*)(ws + take((size_t)25 * 512 * 2));
  _Float16* WRH   = (_Float16*)(ws + take((size_t)25 * 2 * 8 * 512 * 2));
  _Float16* WHRH  = (_Float16*)(ws + take((size_t)10 * 8 * 512 * 2));
  float*    BIASRT= (float*)   (ws + take(128 * 4));

  dim3 blk(256);
  pack_nhwc_kernel<<<dim3(Bn * HWn / 64), blk, 0, stream>>>(inputx, hid, XCAT);
  {
    int total = 25 * 4 * 2 * 512 + 25 * 512 + 25 * 2 * 8 * 512 + 10 * 8 * 512 + 128;
    pack_weights_kernel<<<dim3((total + 255) / 256), blk, 0, stream>>>(
        gxw, ghw, flw, wxr, wxh, whr, whh, whrb, whhb,
        WCAT1, WFLOW, WRH, WHRH, BIASRT);
  }
  conv_f_kernel<<<dim3(Bn * Hn), blk, 0, stream>>>(XCAT, WCAT1, gxb, ghb, F16);
  conv_flow_kernel<<<dim3(Bn * Hn), blk, 0, stream>>>(F16, WFLOW, flb, FLOWS);
  warp_gemm_kernel<<<dim3(Bn * Hn), blk, 0, stream>>>(XCAT, FLOWS, WHRH, BIASRT, TEMPS);
  conv_final_kernel<<<dim3(Bn * Hn), blk, 0, stream>>>(XCAT, WRH, wxrb, wxhb, TEMPS, out);
}
